// MultiHeadAttention_67018669687426
// MI455X (gfx1250) — matrix operand order repaired and checked by reading, not compiled, not run on hardware
//
#include <hip/hip_runtime.h>
#include <hip/hip_bf16.h>

typedef __attribute__((ext_vector_type(16))) _Float16 v16h;
typedef __attribute__((ext_vector_type(8)))  _Float16 v8h;
typedef __attribute__((ext_vector_type(8)))  float    v8f;

#define S_LEN   2048
#define DMODEL  1024
#define NHEADS  16
#define DK      64
#define BATCH   2
#define SC_PITCH 2052   // 2048 + 4 pad floats -> rotates LDS banks per row

#if defined(__has_builtin)
#if __has_builtin(__builtin_amdgcn_tensor_load_to_lds) && __has_builtin(__builtin_amdgcn_s_wait_tensorcnt)
#define HAVE_TDM 1
#endif
#endif

typedef unsigned int u32x4 __attribute__((ext_vector_type(4)));
typedef int          i32x8 __attribute__((ext_vector_type(8)));
typedef int          i32x4 __attribute__((ext_vector_type(4)));

// ---------------------------------------------------------------------------
// Projection GEMM: out = X @ W^T + bias   (unchanged; clean WMMA codegen)
// MODE 0: f16 out, head-split [bh][s][d]        (q, k)
// MODE 1: f16 out, head-split transposed [bh][d][s] (v)
// MODE 2: f32 out, plain [m][n]                 (final projection)
// ---------------------------------------------------------------------------
template<int MODE>
__global__ __launch_bounds__(256) void proj_kernel(
    const float* __restrict__ X, const float* __restrict__ W,
    const float* __restrict__ bias, void* __restrict__ outp)
{
    const int lane = threadIdx.x;
    const int wave = threadIdx.y;
    const int tile = blockIdx.x * 8 + wave;  // 16384 tiles
    const int tm   = tile >> 6;
    const int tn   = tile & 63;
    const int hi   = lane >> 4;
    const int l15  = lane & 15;
    const int rowA = tm * 16 + l15;
    const int colB = tn * 16 + l15;

    const float* xrow = X + (size_t)rowA * DMODEL;
    const float* wrow = W + (size_t)colB * DMODEL;

    v8f c = {};
    for (int k0 = 0; k0 < DMODEL; k0 += 32) {
        v16h a, b;
        {
            const float4* p0 = (const float4*)(xrow + k0 + hi * 8);
            const float4* p1 = (const float4*)(xrow + k0 + 16 + hi * 8);
            float4 x0 = p0[0], x1 = p0[1], x2 = p1[0], x3 = p1[1];
            a[0]=(_Float16)x0.x; a[1]=(_Float16)x0.y; a[2]=(_Float16)x0.z; a[3]=(_Float16)x0.w;
            a[4]=(_Float16)x1.x; a[5]=(_Float16)x1.y; a[6]=(_Float16)x1.z; a[7]=(_Float16)x1.w;
            a[8]=(_Float16)x2.x; a[9]=(_Float16)x2.y; a[10]=(_Float16)x2.z; a[11]=(_Float16)x2.w;
            a[12]=(_Float16)x3.x; a[13]=(_Float16)x3.y; a[14]=(_Float16)x3.z; a[15]=(_Float16)x3.w;
        }
        {
            const float4* p0 = (const float4*)(wrow + k0 + hi * 8);
            const float4* p1 = (const float4*)(wrow + k0 + 16 + hi * 8);
            float4 w0 = p0[0], w1 = p0[1], w2 = p1[0], w3 = p1[1];
            b[0]=(_Float16)w0.x; b[1]=(_Float16)w0.y; b[2]=(_Float16)w0.z; b[3]=(_Float16)w0.w;
            b[4]=(_Float16)w1.x; b[5]=(_Float16)w1.y; b[6]=(_Float16)w1.z; b[7]=(_Float16)w1.w;
            b[8]=(_Float16)w2.x; b[9]=(_Float16)w2.y; b[10]=(_Float16)w2.z; b[11]=(_Float16)w2.w;
            b[12]=(_Float16)w3.x; b[13]=(_Float16)w3.y; b[14]=(_Float16)w3.z; b[15]=(_Float16)w3.w;
        }
        c = __builtin_amdgcn_wmma_f32_16x16x32_f16(false, a, false, b, (short)0, c, false, false);
    }

    const float bv = bias[colB];
    #pragma unroll
    for (int i = 0; i < 8; ++i) {
        const int m = tm * 16 + i + hi * 8;
        const int n = colB;
        const float v = c[i] + bv;
        if (MODE == 2) {
            ((float*)outp)[(size_t)m * DMODEL + n] = v;
        } else {
            const int bb = m >> 11, s = m & (S_LEN - 1);
            const int h  = n >> 6,  d = n & (DK - 1);
            if (MODE == 0)
                ((_Float16*)outp)[(((size_t)(bb * NHEADS + h) * S_LEN + s) << 6) + d] = (_Float16)v;
            else
                ((_Float16*)outp)[(((size_t)(bb * NHEADS + h) * DK + d) << 11) + s] = (_Float16)v;
        }
    }
}

// ---------------------------------------------------------------------------
// Fused attention: per (bh, 16-query stripe):
//   phase 0: stage q tile (16x64 f16) into LDS (TDM if available)
//   phase 1: scores stripe 16x2048 f32 into LDS via WMMA (8 waves x 16 k-tiles)
//   phase 2: in-LDS softmax (2 rows/wave, shfl reductions), probs -> HBM once
//   phase 3: ctx = probs @ V via WMMA, A-fragments read from LDS,
//            K split across 2 wave-groups, LDS partial reduce, ctx -> ws
// ---------------------------------------------------------------------------
__global__ __launch_bounds__(256) void attn_fused_kernel(
    const _Float16* __restrict__ q16, const _Float16* __restrict__ k16,
    const _Float16* __restrict__ vT16,
    float* __restrict__ probs, float* __restrict__ ctx)
{
    __shared__ float    sc[16][SC_PITCH];     // 131 KB score/prob stripe
    __shared__ _Float16 qlds[16 * DK];        // 2 KB q tile
    __shared__ float    redc[4][8][32];       // 4 KB ctx partial reduce

    const int lane = threadIdx.x;
    const int wave = threadIdx.y;
    const int bh   = blockIdx.x >> 7;         // 0..31
    const int tq   = blockIdx.x & 127;
    const int q0   = tq * 16;
    const int hi   = lane >> 4;
    const int l15  = lane & 15;

    // ---- phase 0: stage q tile into LDS --------------------------------
    const _Float16* qsrc = q16 + ((size_t)bh * S_LEN + q0) * DK;
#if HAVE_TDM
    if (wave == 0) {
        const unsigned long long ga = (unsigned long long)(uintptr_t)qsrc;
        const unsigned int lds_addr = (unsigned int)(uintptr_t)&qlds[0];
        u32x4 g0 = { 1u,                                   // count=1, user desc
                     lds_addr,                             // lds_addr [63:32]
                     (unsigned int)(ga & 0xFFFFFFFFu),     // global_addr [95:64]
                     (unsigned int)((ga >> 32) & 0x1FFFFFFu) | 0x80000000u }; // [120:96] + type=2
        i32x8 g1 = { (int)0x00010000,       // data_size=1 (2B), no multicast
                     (int)(64u  << 16),     // tensor_dim0 = 64
                     (int)(2048u << 16),    // tensor_dim1 = 2048 (low16 @ bit80)
                     (int)(64u  << 16),     // tile_dim0 = 64 (@ bit112)
                     (int)16,               // tile_dim1 = 16
                     (int)64,               // tensor_dim0_stride = 64
                     0, 0 };
        i32x4 gz4 = { 0, 0, 0, 0 };
        i32x8 gz8 = { 0, 0, 0, 0, 0, 0, 0, 0 };
        __builtin_amdgcn_tensor_load_to_lds(g0, g1, gz4, gz4, gz8, 0);
        __builtin_amdgcn_s_wait_tensorcnt((short)0);
    }
#else
    {   // cooperative copy: 2048 bytes, 256 threads x 8B
        const int tid = wave * 32 + lane;
        ((unsigned long long*)qlds)[tid] = ((const unsigned long long*)qsrc)[tid];
    }
#endif
    __syncthreads();

    // A fragments for the stripe (from LDS), reused for all 128 key tiles
    v16h afrag[2];
    #pragma unroll
    for (int kk = 0; kk < 2; ++kk) {
        const int k0 = kk * 32;
        v8h a0 = *(const v8h*)(&qlds[l15 * DK + k0 + hi * 8]);
        v8h a1 = *(const v8h*)(&qlds[l15 * DK + k0 + 16 + hi * 8]);
        #pragma unroll
        for (int j = 0; j < 8; ++j) { afrag[kk][j] = a0[j]; afrag[kk][8 + j] = a1[j]; }
    }

    // ---- phase 1: scores stripe into LDS -------------------------------
    for (int t = 0; t < 16; ++t) {
        const int tk = wave + t * 8;         // 0..127
        const _Float16* krow = k16 + ((size_t)bh * S_LEN + tk * 16 + l15) * DK;
        v8f c = {};
        #pragma unroll
        for (int kk = 0; kk < 2; ++kk) {
            const int k0 = kk * 32;
            v16h b;
            v8h b0 = *(const v8h*)(krow + k0 + hi * 8);
            v8h b1 = *(const v8h*)(krow + k0 + 16 + hi * 8);
            #pragma unroll
            for (int j = 0; j < 8; ++j) { b[j] = b0[j]; b[8 + j] = b1[j]; }
            c = __builtin_amdgcn_wmma_f32_16x16x32_f16(false, afrag[kk], false, b, (short)0, c, false, false);
        }
        const int kc = tk * 16 + l15;
        #pragma unroll
        for (int i = 0; i < 8; ++i)
            sc[i + hi * 8][kc] = c[i] * 0.125f;   // 1/sqrt(64)
    }
    __syncthreads();

    // ---- phase 2: softmax over each of the 16 rows (2 rows per wave) ---
    #pragma unroll
    for (int rr = 0; rr < 2; ++rr) {
        const int r = wave * 2 + rr;
        float* row = &sc[r][0];
        float m = -3.402823466e38f;
        for (int j = lane; j < S_LEN; j += 32) m = fmaxf(m, row[j]);
        #pragma unroll
        for (int o = 16; o > 0; o >>= 1) m = fmaxf(m, __shfl_xor(m, o, 32));
        float sum = 0.f;
        for (int j = lane; j < S_LEN; j += 32) {
            float e = __expf(row[j] - m);
            row[j] = e;
            sum += e;
        }
        #pragma unroll
        for (int o = 16; o > 0; o >>= 1) sum += __shfl_xor(sum, o, 32);
        const float inv = 1.0f / sum;
        float* gout = probs + ((size_t)bh * S_LEN + q0 + r) * S_LEN;
        for (int j = lane; j < S_LEN; j += 32) {
            const float p = row[j] * inv;
            row[j] = p;
            gout[j] = p;
        }
    }
    __syncthreads();

    // ---- phase 3: ctx = probs @ V  (A from LDS, B from global) ---------
    const int tn = wave & 3;                  // dk column tile
    const int kh = wave >> 2;                 // K half (0: 0..1023, 1: 1024..2047)
    const _Float16* vrow = vT16 + ((size_t)bh * DK + tn * 16 + l15) * S_LEN;
    const float*    prow = &sc[l15][0];

    v8f c = {};
    for (int k0 = kh * 1024; k0 < kh * 1024 + 1024; k0 += 32) {
        v16h a, b;
        {
            const float4* p0 = (const float4*)(prow + k0 + hi * 8);
            const float4* p1 = (const float4*)(prow + k0 + 16 + hi * 8);
            float4 x0 = p0[0], x1 = p0[1], x2 = p1[0], x3 = p1[1];
            a[0]=(_Float16)x0.x; a[1]=(_Float16)x0.y; a[2]=(_Float16)x0.z; a[3]=(_Float16)x0.w;
            a[4]=(_Float16)x1.x; a[5]=(_Float16)x1.y; a[6]=(_Float16)x1.z; a[7]=(_Float16)x1.w;
            a[8]=(_Float16)x2.x; a[9]=(_Float16)x2.y; a[10]=(_Float16)x2.z; a[11]=(_Float16)x2.w;
            a[12]=(_Float16)x3.x; a[13]=(_Float16)x3.y; a[14]=(_Float16)x3.z; a[15]=(_Float16)x3.w;
        }
        {
            v8h b0 = *(const v8h*)(vrow + k0 + hi * 8);
            v8h b1 = *(const v8h*)(vrow + k0 + 16 + hi * 8);
            #pragma unroll
            for (int j = 0; j < 8; ++j) { b[j] = b0[j]; b[8 + j] = b1[j]; }
        }
        c = __builtin_amdgcn_wmma_f32_16x16x32_f16(false, a, false, b, (short)0, c, false, false);
    }

    if (kh == 1) {
        #pragma unroll
        for (int i = 0; i < 8; ++i) redc[tn][i][lane] = c[i];
    }
    __syncthreads();
    if (kh == 0) {
        const int bb = bh >> 4, h = bh & 15;
        const int d  = tn * 16 + l15;
        #pragma unroll
        for (int i = 0; i < 8; ++i) {
            const float v = c[i] + redc[tn][i][lane];
            const int qr = q0 + i + hi * 8;
            ctx[((size_t)bb * S_LEN + qr) * DMODEL + h * DK + d] = v;
        }
    }
}

// ---------------------------------------------------------------------------
extern "C" void kernel_launch(void* const* d_in, const int* in_sizes, int n_in,
                              void* d_out, int out_size, void* d_ws, size_t ws_size,
                              hipStream_t stream) {
    const float* Q  = (const float*)d_in[0];
    const float* K  = (const float*)d_in[1];
    const float* V  = (const float*)d_in[2];
    const float* Wq = (const float*)d_in[3]; const float* bq = (const float*)d_in[4];
    const float* Wk = (const float*)d_in[5]; const float* bk = (const float*)d_in[6];
    const float* Wv = (const float*)d_in[7]; const float* bv = (const float*)d_in[8];
    const float* Wo = (const float*)d_in[9]; const float* bo = (const float*)d_in[10];

    float* probs     = (float*)d_out;                        // [2,16,2048,2048]
    float* final_out = (float*)d_out + 134217728ull;         // [2,2048,1024]

    char* ws = (char*)d_ws;
    _Float16* q16  = (_Float16*)(ws);                        //  8 MB
    _Float16* k16  = (_Float16*)(ws + 8388608);              //  8 MB
    _Float16* vT16 = (_Float16*)(ws + 16777216);             //  8 MB
    float*    ctx  = (float*)   (ws + 25165824);             // 16 MB

    dim3 blk(32, 8);
    proj_kernel<0><<<2048, blk, 0, stream>>>(Q, Wq, bq, (void*)q16);
    proj_kernel<0><<<2048, blk, 0, stream>>>(K, Wk, bk, (void*)k16);
    proj_kernel<1><<<2048, blk, 0, stream>>>(V, Wv, bv, (void*)vT16);
    attn_fused_kernel<<<4096, blk, 0, stream>>>(q16, k16, vT16, probs, ctx);
    proj_kernel<2><<<2048, blk, 0, stream>>>(ctx, Wo, bo, (void*)final_out);
}
